// DownConv_40544491274254
// MI455X (gfx1250) — matrix operand order repaired and checked by reading, not compiled, not run on hardware
//
#include <hip/hip_runtime.h>

// ---------------------------------------------------------------------------
// MI455X (gfx1250) tri-plane network.
// GEMM-shaped work -> v_wmma_f32_16x16x32_f16, f16 operands staged into LDS
// with GLOBAL_LOAD_ASYNC_TO_LDS_B128 (ASYNCcnt-tracked, double buffered).
// ---------------------------------------------------------------------------

typedef _Float16 v16h __attribute__((ext_vector_type(16)));
typedef _Float16 v8h  __attribute__((ext_vector_type(8)));
typedef float    v8f  __attribute__((ext_vector_type(8)));

#define WMMA(a, b, c) \
  __builtin_amdgcn_wmma_f32_16x16x32_f16(false, (a), false, (b), (short)0, (c), false, false)

static __device__ __forceinline__ v16h frag_join(const _Float16* p0, const _Float16* p1) {
  v8h lo = *(const v8h*)p0;
  v8h hi = *(const v8h*)p1;
  return __builtin_shufflevector(lo, hi, 0,1,2,3,4,5,6,7,8,9,10,11,12,13,14,15);
}
// A fragment (16x32 f16): fixed row, runs at k=half*8 and 16+half*8
static __device__ __forceinline__ v16h frag_a(const _Float16* row, int half) {
  return frag_join(row + half * 8, row + 16 + half * 8);
}
// B fragment (32x16 f16): fixed col, runs at k=half*8 and 16+half*8
static __device__ __forceinline__ v16h frag_b(const _Float16* col, int half) {
  return frag_join(col + half * 8, col + 16 + half * 8);
}

// low 32 bits of a generic pointer to __shared__ = wave-relative LDS address
static __device__ __forceinline__ unsigned lds_addr(const void* p) {
  return (unsigned)(uintptr_t)p;
}
// async copy 16B global -> LDS (per lane), tracked with ASYNCcnt
static __device__ __forceinline__ void async_cp16(unsigned lds, const void* g) {
  asm volatile("global_load_async_to_lds_b128 %0, %1, off"
               :: "v"(lds), "v"(g) : "memory");
}
static __device__ __forceinline__ void async_wait(int n) {
  if (n == 0) asm volatile("s_wait_asynccnt 0" ::: "memory");
  else        asm volatile("s_wait_asynccnt 2" ::: "memory");
}

// ---------------------------------------------------------------------------
// Token GEMM: Y[M,F] = act(X[M,K] @ W[F,K]^T + bias) (+ res), X/W f16.
// 256 threads (8 waves), 64x64 tile, K step 32, double-buffered async LDS.
// ---------------------------------------------------------------------------
__global__ __launch_bounds__(256) void gemm_kernel(
    const _Float16* __restrict__ X, const _Float16* __restrict__ Wm,
    const float* __restrict__ bias, const float* __restrict__ res,
    float* __restrict__ Y, _Float16* __restrict__ Yh,
    int M, int K, int F, int relu) {
  __shared__ alignas(32) _Float16 Xs[2][64][32];
  __shared__ alignas(32) _Float16 Ws[2][64][32];
  const int tid  = threadIdx.x;
  const int m0   = blockIdx.x * 64;
  const int n0   = blockIdx.y * 64;
  const int w    = tid >> 5, lane = tid & 31;
  const int wy   = w >> 1,  wx  = w & 1;
  const int half = lane >> 4, l16 = lane & 15;
  const int srow = tid >> 2, scol = (tid & 3) * 8;

  const _Float16* xrow = X  + (size_t)(m0 + srow) * K + scol;
  const _Float16* wrow = Wm + (size_t)(n0 + srow) * K + scol;
  const unsigned lx0 = lds_addr(&Xs[0][srow][scol]);
  const unsigned lx1 = lds_addr(&Xs[1][srow][scol]);
  const unsigned lw0 = lds_addr(&Ws[0][srow][scol]);
  const unsigned lw1 = lds_addr(&Ws[1][srow][scol]);

  v8f acc0 = {0,0,0,0,0,0,0,0};
  v8f acc1 = {0,0,0,0,0,0,0,0};

  async_cp16(lx0, xrow);
  async_cp16(lw0, wrow);
  const int nsteps = K >> 5;
  for (int i = 0; i < nsteps; ++i) {
    if (i + 1 < nsteps) {
      async_cp16((i & 1) ? lx0 : lx1, xrow + (i + 1) * 32);
      async_cp16((i & 1) ? lw0 : lw1, wrow + (i + 1) * 32);
      async_wait(2);   // oldest 2 (current tile) complete; next tile in flight
    } else {
      async_wait(0);
    }
    __syncthreads();
    const _Float16 (*Xc)[32] = Xs[i & 1];
    const _Float16 (*Wc)[32] = Ws[i & 1];
    v16h a  = frag_a(&Xc[wy * 16 + l16][0], half);
    v16h b0 = frag_b(&Wc[wx * 32 + l16][0], half);
    v16h b1 = frag_b(&Wc[wx * 32 + 16 + l16][0], half);
    acc0 = WMMA(a, b0, acc0);
    acc1 = WMMA(a, b1, acc1);
    __syncthreads();
  }

  const int mrow = m0 + wy * 16 + half * 8;
#pragma unroll
  for (int r = 0; r < 8; ++r) {
    int m = mrow + r;
#pragma unroll
    for (int s = 0; s < 2; ++s) {
      int n = n0 + wx * 32 + s * 16 + l16;
      float v = (s ? acc1[r] : acc0[r]) + bias[n];
      if (relu) v = fmaxf(v, 0.f);
      size_t off = (size_t)m * F + n;
      if (res) v += res[off];
      Y[off] = v;
      if (Yh) Yh[off] = (_Float16)v;
    }
  }
}

// ---------------------------------------------------------------------------
// 3x3 conv, pad 1, C=128, NCHW (6x128x128), f16 input, packed f16 weights
// [tap][O][C] (async-staged). A = weights, B = pixels.
// grid = (xblk 2, y 128, img*2+oblk 12)
// ---------------------------------------------------------------------------
__global__ __launch_bounds__(256) void conv3x3_kernel(
    const _Float16* __restrict__ in, const _Float16* __restrict__ wpk,
    const float* __restrict__ bias, const float* __restrict__ res,
    float* __restrict__ out, float* __restrict__ out2,
    _Float16* __restrict__ outh, int relu) {
  __shared__ alignas(32) _Float16 Ws[64][32];  // [o_local][c_local]
  __shared__ alignas(32) _Float16 Xs[64][32];  // [px_local][c_local]
  const int tid  = threadIdx.x;
  const int x0   = blockIdx.x * 64;
  const int y    = blockIdx.y;
  const int img  = blockIdx.z >> 1;
  const int o0   = (blockIdx.z & 1) * 64;
  const int w    = tid >> 5, lane = tid & 31;
  const int wo   = w >> 1,  wpx = w & 1;
  const int half = lane >> 4, l16 = lane & 15;
  const int srow = tid >> 2, scol = (tid & 3) * 8;
  const unsigned lw = lds_addr(&Ws[srow][scol]);

  v8f acc0 = {0,0,0,0,0,0,0,0};
  v8f acc1 = {0,0,0,0,0,0,0,0};

  for (int tap = 0; tap < 9; ++tap) {
    const int ky = tap / 3, kx = tap % 3;
    const int yy = y + ky - 1;
    const bool yok = (yy >= 0) && (yy < 128);
    const int xx = x0 + srow + kx - 1;
    const bool ok = yok && (xx >= 0) && (xx < 128);
    for (int cb = 0; cb < 128; cb += 32) {
      // weights: contiguous -> async to LDS
      async_cp16(lw, wpk + ((size_t)tap * 128 + o0 + srow) * 128 + cb + scol);
      // input patch: padded -> guarded scalar f16 loads
      const _Float16* iptr = in + ((size_t)img * 128 + cb + scol) * 16384 + (size_t)yy * 128 + xx;
#pragma unroll
      for (int j = 0; j < 8; ++j)
        Xs[srow][scol + j] = ok ? iptr[(size_t)j * 16384] : (_Float16)0.f;
      async_wait(0);
      __syncthreads();
      v16h a  = frag_a(&Ws[wo * 16 + l16][0], half);
      v16h b0 = frag_b(&Xs[wpx * 32 + l16][0], half);
      v16h b1 = frag_b(&Xs[wpx * 32 + 16 + l16][0], half);
      acc0 = WMMA(a, b0, acc0);
      acc1 = WMMA(a, b1, acc1);
      __syncthreads();
    }
  }

#pragma unroll
  for (int r = 0; r < 8; ++r) {
    const int o = o0 + wo * 16 + half * 8 + r;
    const float bv = bias[o];
#pragma unroll
    for (int s = 0; s < 2; ++s) {
      int xg = x0 + wpx * 32 + s * 16 + l16;
      size_t off = ((size_t)img * 128 + o) * 16384 + (size_t)y * 128 + xg;
      float v = (s ? acc1[r] : acc0[r]) + bv;
      if (relu) v = fmaxf(v, 0.f);
      if (res)  v += res[off];
      if (out)  out[off]  = v;
      if (out2) out2[off] = v;
      if (outh) outh[off] = (_Float16)v;
    }
  }
}

// ---------------------------------------------------------------------------
// conversions / weight packing
// ---------------------------------------------------------------------------
__global__ __launch_bounds__(256) void cvt_kernel(
    const float* __restrict__ src, _Float16* __restrict__ dst, size_t n) {
  size_t i = (size_t)blockIdx.x * blockDim.x + threadIdx.x;
  if (i < n) dst[i] = (_Float16)src[i];
}

// (O,C,3,3) fp32 -> [tap][O][C] f16
__global__ __launch_bounds__(256) void pack_convw_kernel(
    const float* __restrict__ src, _Float16* __restrict__ dst) {
  int i = blockIdx.x * blockDim.x + threadIdx.x;
  if (i >= 9 * 128 * 128) return;
  int tap = i / 16384;
  int rem = i & 16383;
  int o = rem >> 7, c = rem & 127;
  dst[i] = (_Float16)src[(size_t)(o * 128 + c) * 9 + tap];
}

// ---------------------------------------------------------------------------
// Per-(img,ch) spatial means: mean2[img][c][x] (over y), mean3[img][c][y] (over x)
// ---------------------------------------------------------------------------
__global__ __launch_bounds__(256) void mean_axes_kernel(
    const float* __restrict__ h, float* __restrict__ mean2, float* __restrict__ mean3) {
  int i = blockIdx.x * blockDim.x + threadIdx.x;
  if (i >= 6 * 128 * 128) return;
  int imgc = i >> 7;
  int t = i & 127;
  const float* base = h + (size_t)imgc * 16384;
  float s2 = 0.f, s3 = 0.f;
  for (int j = 0; j < 128; ++j) {
    s2 += base[j * 128 + t];
    s3 += base[t * 128 + j];
  }
  mean2[i] = s2 * (1.f / 128.f);
  mean3[i] = s3 * (1.f / 128.f);
}

// ---------------------------------------------------------------------------
// Rodin 1x1 conv (K=384) with fused mean-broadcast input selection (fp32 in).
// ---------------------------------------------------------------------------
__global__ __launch_bounds__(256) void rodin_kernel(
    const float* __restrict__ xafter, const float* __restrict__ mean2,
    const float* __restrict__ mean3, const float* __restrict__ rw,
    const float* __restrict__ rb, float* __restrict__ xr, int Bc) {
  __shared__ alignas(32) _Float16 Ws[64][32];
  __shared__ alignas(32) _Float16 Xs[64][32];
  const int tid  = threadIdx.x;
  const int x0   = blockIdx.x * 64;
  const int y    = blockIdx.y;
  const int pb   = blockIdx.z >> 1;
  const int o0   = (blockIdx.z & 1) * 64;
  const int p    = pb / Bc, b = pb % Bc;
  const int w    = tid >> 5, lane = tid & 31;
  const int wo   = w >> 1,  wpx = w & 1;
  const int half = lane >> 4, l16 = lane & 15;
  const int srow = tid >> 2, scol = (tid & 3) * 8;

  int segmode[3], segimg[3];
  segmode[0] = 0;
  segimg[0]  = (p == 0) ? b : (p == 1) ? (Bc + b) : (2 * Bc + b);
  if (p == 0)      { segmode[1] = 1; segimg[1] = Bc + b;     segmode[2] = 1; segimg[2] = 2 * Bc + b; }
  else if (p == 1) { segmode[1] = 1; segimg[1] = b;          segmode[2] = 2; segimg[2] = 2 * Bc + b; }
  else             { segmode[1] = 2; segimg[1] = b;          segmode[2] = 2; segimg[2] = Bc + b; }

  v8f acc0 = {0,0,0,0,0,0,0,0};
  v8f acc1 = {0,0,0,0,0,0,0,0};

  for (int k0 = 0; k0 < 384; k0 += 32) {
    const int seg = k0 >> 7, kc0 = k0 & 127;
    const int mode = segmode[seg], img = segimg[seg];
    const float* wptr = rw + ((size_t)p * 128 + o0 + srow) * 384 + k0 + scol;
#pragma unroll
    for (int j = 0; j < 8; ++j) Ws[srow][scol + j] = (_Float16)wptr[j];
    const int xg = x0 + srow;
#pragma unroll
    for (int j = 0; j < 8; ++j) {
      const int kc = kc0 + scol + j;
      float v;
      if (mode == 0)      v = xafter[((size_t)img * 128 + kc) * 16384 + (size_t)y * 128 + xg];
      else if (mode == 1) v = mean2[((size_t)img * 128 + kc) * 128 + xg];
      else                v = mean3[((size_t)img * 128 + kc) * 128 + y];
      Xs[srow][scol + j] = (_Float16)v;
    }
    __syncthreads();
    v16h a  = frag_a(&Ws[wo * 16 + l16][0], half);
    v16h b0 = frag_b(&Xs[wpx * 32 + l16][0], half);
    v16h b1 = frag_b(&Xs[wpx * 32 + 16 + l16][0], half);
    acc0 = WMMA(a, b0, acc0);
    acc1 = WMMA(a, b1, acc1);
    __syncthreads();
  }

#pragma unroll
  for (int r = 0; r < 8; ++r) {
    const int o = o0 + wo * 16 + half * 8 + r;
    const float bv = rb[p * 128 + o];
#pragma unroll
    for (int s = 0; s < 2; ++s) {
      int xg = x0 + wpx * 32 + s * 16 + l16;
      xr[((size_t)pb * 128 + o) * 16384 + (size_t)y * 128 + xg] = (s ? acc1[r] : acc0[r]) + bv;
    }
  }
}

// ---------------------------------------------------------------------------
// Bilinear tri-plane sampling -> f16 tokens (consumed by fc1 GEMM)
// ---------------------------------------------------------------------------
__global__ __launch_bounds__(128) void bilinear_kernel(
    const float* __restrict__ xr, const float* __restrict__ xyz,
    _Float16* __restrict__ out, int Bc, int Nc) {
  const int tok = blockIdx.x;
  const int b = tok / Nc;
  const float* xp = xyz + (size_t)tok * 3;
  float pn0 = xp[0] * (1.f / 1.1f) + 0.5f;
  float pn1 = xp[1] * (1.f / 1.1f) + 0.5f;
  float pn2 = xp[2] * (1.f / 1.1f) + 0.5f;
  const int ch = threadIdx.x;
  float accv = 0.f;
#pragma unroll
  for (int p = 0; p < 3; ++p) {
    float u = ((p == 2) ? pn1 : pn0) * 127.f;
    float v = ((p == 0) ? pn1 : pn2) * 127.f;
    float uf = floorf(u), vf = floorf(v);
    float wu = u - uf, wv = v - vf;
    int u0 = min(max((int)uf, 0), 127), v0 = min(max((int)vf, 0), 127);
    int u1 = min(u0 + 1, 127), v1 = min(v0 + 1, 127);
    const float* pl = xr + (((size_t)p * Bc + b) * 128 + ch) * 16384;
    float a00 = pl[v0 * 128 + u0], a01 = pl[v0 * 128 + u1];
    float a10 = pl[v1 * 128 + u0], a11 = pl[v1 * 128 + u1];
    accv += a00 * (1.f - wu) * (1.f - wv) + a01 * wu * (1.f - wv)
          + a10 * (1.f - wu) * wv + a11 * wu * wv;
  }
  out[(size_t)tok * 128 + ch] = (_Float16)accv;
}

// ---------------------------------------------------------------------------
// Row gather (fp32): out[b][n][:] = in[b][idx[b][n]][:]
// ---------------------------------------------------------------------------
__global__ __launch_bounds__(128) void gather_kernel(
    const float* __restrict__ in, const int* __restrict__ idx,
    float* __restrict__ out, int Nc, int Dc) {
  const int tok = blockIdx.x;
  const int b = tok / Nc;
  const int src = idx[tok];
  const float* ip = in + ((size_t)b * Nc + src) * Dc;
  float* op = out + (size_t)tok * Dc;
  for (int c = threadIdx.x; c < Dc; c += blockDim.x) op[c] = ip[c];
}

// ---------------------------------------------------------------------------
// LayerNorm(D=128): wave32 per token, f32 in -> f16 out (GEMM operand)
// ---------------------------------------------------------------------------
__global__ __launch_bounds__(256) void ln_kernel(
    const float* __restrict__ x, const float* __restrict__ g,
    const float* __restrict__ bb, _Float16* __restrict__ y, int ntok) {
  const int tok = blockIdx.x * (blockDim.x >> 5) + (threadIdx.x >> 5);
  const int lane = threadIdx.x & 31;
  if (tok >= ntok) return;
  const float* xp = x + (size_t)tok * 128;
  float v0 = xp[lane], v1 = xp[lane + 32], v2 = xp[lane + 64], v3 = xp[lane + 96];
  float s = v0 + v1 + v2 + v3;
#pragma unroll
  for (int off = 16; off > 0; off >>= 1) s += __shfl_xor(s, off, 32);
  const float mean = s * (1.f / 128.f);
  float d0 = v0 - mean, d1 = v1 - mean, d2 = v2 - mean, d3 = v3 - mean;
  float sq = d0 * d0 + d1 * d1 + d2 * d2 + d3 * d3;
#pragma unroll
  for (int off = 16; off > 0; off >>= 1) sq += __shfl_xor(sq, off, 32);
  const float inv = rsqrtf(sq * (1.f / 128.f) + 1e-5f);
  _Float16* yp = y + (size_t)tok * 128;
  yp[lane]      = (_Float16)(d0 * inv * g[lane]      + bb[lane]);
  yp[lane + 32] = (_Float16)(d1 * inv * g[lane + 32] + bb[lane + 32]);
  yp[lane + 64] = (_Float16)(d2 * inv * g[lane + 64] + bb[lane + 64]);
  yp[lane + 96] = (_Float16)(d3 * inv * g[lane + 96] + bb[lane + 96]);
}

// ---------------------------------------------------------------------------
// Windowed attention: W=125, HD=32, H=4, online softmax; f16 out (proj GEMM)
// ---------------------------------------------------------------------------
__global__ __launch_bounds__(128) void attn_kernel(
    const float* __restrict__ qkv, _Float16* __restrict__ out, int Nc) {
  const int id = blockIdx.x;
  const int head = id & 3;
  const int win = (id >> 2) & 127;
  const int b = id >> 9;
  __shared__ float kbuf[125][32];
  __shared__ float vbuf[125][32];
  const size_t base = ((size_t)b * Nc + (size_t)win * 125) * 384;
  const int hoff = head * 32;
  for (int i = threadIdx.x; i < 125 * 32; i += 128) {
    int t = i >> 5, d = i & 31;
    kbuf[t][d] = qkv[base + (size_t)t * 384 + 128 + hoff + d];
    vbuf[t][d] = qkv[base + (size_t)t * 384 + 256 + hoff + d];
  }
  __syncthreads();
  const int q = threadIdx.x;
  if (q < 125) {
    float qv[32], oa[32];
    const float* qp = qkv + base + (size_t)q * 384 + hoff;
#pragma unroll
    for (int d = 0; d < 32; ++d) { qv[d] = qp[d]; oa[d] = 0.f; }
    float m = -1e30f, l = 0.f;
    const float scale = 0.17677669529663687f;  // 1/sqrt(32)
    for (int j = 0; j < 125; ++j) {
      float s = 0.f;
#pragma unroll
      for (int d = 0; d < 32; ++d) s += qv[d] * kbuf[j][d];
      s *= scale;
      float mnew = fmaxf(m, s);
      float corr = __expf(m - mnew);
      float pexp = __expf(s - mnew);
      l = l * corr + pexp;
#pragma unroll
      for (int d = 0; d < 32; ++d) oa[d] = oa[d] * corr + pexp * vbuf[j][d];
      m = mnew;
    }
    const float invl = 1.f / l;
    _Float16* op = out + ((size_t)b * Nc + (size_t)win * 125 + q) * 128 + hoff;
#pragma unroll
    for (int d = 0; d < 32; ++d) op[d] = (_Float16)(oa[d] * invl);
  }
}

// ---------------------------------------------------------------------------
// Scatter-mean pieces
// ---------------------------------------------------------------------------
__global__ __launch_bounds__(256) void zero_kernel(float* __restrict__ p, size_t n) {
  size_t i = (size_t)blockIdx.x * blockDim.x + threadIdx.x;
  if (i < n) p[i] = 0.f;
}

__global__ __launch_bounds__(128) void scatter_add_kernel(
    const float* __restrict__ c, const float* __restrict__ xyz,
    float* __restrict__ sums, float* __restrict__ cnts, int Bc, int Nc) {
  const int pt = blockIdx.x;
  const int p = blockIdx.y;
  const int b = pt / Nc;
  const float* xp = xyz + (size_t)pt * 3;
  float pn0 = xp[0] * (1.f / 1.1f) + 0.5f;
  float pn1 = xp[1] * (1.f / 1.1f) + 0.5f;
  float pn2 = xp[2] * (1.f / 1.1f) + 0.5f;
  int g0 = min(max((int)floorf(pn0 * 128.f), 0), 127);
  int g1 = min(max((int)floorf(pn1 * 128.f), 0), 127);
  int g2 = min(max((int)floorf(pn2 * 128.f), 0), 127);
  int flat = (p == 0) ? (g1 * 128 + g0) : (p == 1) ? (g2 * 128 + g0) : (g2 * 128 + g1);
  float* sbase = sums + (((size_t)p * Bc + b) * 16384 + flat) * 128;
  const float* cp = c + (size_t)pt * 128;
  for (int ch = threadIdx.x; ch < 128; ch += blockDim.x) atomicAdd(&sbase[ch], cp[ch]);
  if (threadIdx.x == 0) atomicAdd(&cnts[((size_t)p * Bc + b) * 16384 + flat], 1.f);
}

// sums [pb][flat][ch] -> hcat f16 [pb][ch][flat] / max(count,1)
__global__ __launch_bounds__(256) void scatter_fin_kernel(
    const float* __restrict__ sums, const float* __restrict__ cnts,
    _Float16* __restrict__ hcat) {
  size_t i = (size_t)blockIdx.x * blockDim.x + threadIdx.x;
  const size_t total = (size_t)6 * 16384 * 128;
  if (i >= total) return;
  size_t ch = i & 127, rest = i >> 7;
  size_t flat = rest & 16383, pb = rest >> 14;
  float cnt = cnts[pb * 16384 + flat];
  hcat[(pb * 128 + ch) * 16384 + flat] = (_Float16)(sums[i] / fmaxf(cnt, 1.f));
}

// ---------------------------------------------------------------------------
// Host orchestration
// ---------------------------------------------------------------------------
extern "C" void kernel_launch(void* const* d_in, const int* in_sizes, int n_in,
                              void* d_out, int out_size, void* d_ws, size_t ws_size,
                              hipStream_t stream) {
  (void)in_sizes; (void)n_in; (void)out_size; (void)ws_size;
  const int Bc = 2, Nc = 16000;
  const size_t PLANE = (size_t)6 * 128 * 128 * 128;   // 12,582,912
  const size_t CSZ = (size_t)Bc * Nc * 128;           // 4,096,000
  const size_t QSZ = (size_t)Bc * Nc * 384;

  const float* x       = (const float*)d_in[0];
  const float* c_res   = (const float*)d_in[1];
  const float* xyz     = (const float*)d_in[2];
  /* d_in[3] = x_after_conv : unused by the forward pass */
  const int*   indices     = (const int*)d_in[4];
  const int*   indices_inv = (const int*)d_in[5];
  const float* conv1_w = (const float*)d_in[6];
  const float* conv1_b = (const float*)d_in[7];
  const float* conv2_w = (const float*)d_in[8];
  const float* conv2_b = (const float*)d_in[9];
  const float* rodin_w = (const float*)d_in[10];
  const float* rodin_b = (const float*)d_in[11];
  const float* fc1_w1  = (const float*)d_in[12];
  const float* fc1_b1  = (const float*)d_in[13];
  const float* fc1_w2  = (const float*)d_in[14];
  const float* fc1_b2  = (const float*)d_in[15];
  const float* fc2_w   = (const float*)d_in[16];
  const float* fc2_b   = (const float*)d_in[17];
  const float* ln1_g   = (const float*)d_in[18];
  const float* ln1_b   = (const float*)d_in[19];
  const float* qkv_w   = (const float*)d_in[20];
  const float* qkv_b   = (const float*)d_in[21];
  const float* proj_w  = (const float*)d_in[22];
  const float* proj_b  = (const float*)d_in[23];
  const float* ln2_g   = (const float*)d_in[24];
  const float* ln2_b   = (const float*)d_in[25];
  const float* mlp_w1  = (const float*)d_in[26];
  const float* mlp_b1  = (const float*)d_in[27];
  const float* mlp_w2  = (const float*)d_in[28];
  const float* mlp_b2  = (const float*)d_in[29];
  const float* fus_w   = (const float*)d_in[30];
  const float* fus_b   = (const float*)d_in[31];

  // output regions: xout | xout | x_after | c
  float* out0 = (float*)d_out;
  float* out1 = out0 + PLANE;
  float* outx = out0 + 2 * PLANE;       // x_after (rodin input)
  float* outc = out0 + 3 * PLANE;       // final c

  // ---- workspace: f32 region then f16 region (both 64-elt aligned) ----
  float* wsf = (float*)d_ws;
  size_t woff = 0;
  auto walloc = [&](size_t n) { float* p = wsf + woff; woff += (n + 63) & ~(size_t)63; return p; };
  float* ssum   = walloc(PLANE);        // scatter sums
  float* xr     = walloc(PLANE);        // rodin planes
  float* qkvb   = walloc(QSZ);          // qkv fp32 (attention input)
  float* cbuf   = walloc(CSZ);
  float* cpbuf  = walloc(CSZ);
  float* y2buf  = walloc(CSZ);
  float* mean2b = walloc((size_t)6 * 128 * 128);
  float* mean3b = walloc((size_t)6 * 128 * 128);
  float* cnts   = walloc((size_t)6 * 16384);

  _Float16* hb = (_Float16*)(wsf + woff);
  size_t hoff = 0;
  auto halloc = [&](size_t n) { _Float16* p = hb + hoff; hoff += (n + 63) & ~(size_t)63; return p; };
  _Float16* xh      = halloc(PLANE);    // x in f16
  _Float16* h1h     = halloc(PLANE);    // conv1 output (f16 only)
  _Float16* hcath   = halloc(PLANE);    // scatter-mean result (f16 only)
  _Float16* qkvbh   = halloc(QSZ);      // f16 hidden (fc1 / mlp1)
  _Float16* ybufh   = halloc(CSZ);      // bilinear/LN/attention f16 outputs
  _Float16* cresh   = halloc(CSZ);
  _Float16* cw1h    = halloc((size_t)9 * 128 * 128);
  _Float16* cw2h    = halloc((size_t)9 * 128 * 128);
  _Float16* fwh     = halloc((size_t)9 * 128 * 128);
  _Float16* w11h    = halloc((size_t)256 * 128);
  _Float16* w12h    = halloc((size_t)128 * 256);
  _Float16* w2h     = halloc((size_t)128 * 128);
  _Float16* wqkvh   = halloc((size_t)3 * 384 * 128);
  _Float16* wprojh  = halloc((size_t)3 * 128 * 128);
  _Float16* wmlp1h  = halloc((size_t)3 * 256 * 128);
  _Float16* wmlp2h  = halloc((size_t)3 * 128 * 256);

  auto cvt = [&](const float* s, _Float16* d, size_t n) {
    cvt_kernel<<<(unsigned)((n + 255) / 256), 256, 0, stream>>>(s, d, n);
  };

  // ---- f16 prep ----
  cvt(x, xh, PLANE);
  cvt(c_res, cresh, CSZ);
  cvt(fc1_w1, w11h, (size_t)256 * 128);
  cvt(fc1_w2, w12h, (size_t)128 * 256);
  cvt(fc2_w, w2h, (size_t)128 * 128);
  cvt(qkv_w, wqkvh, (size_t)3 * 384 * 128);
  cvt(proj_w, wprojh, (size_t)3 * 128 * 128);
  cvt(mlp_w1, wmlp1h, (size_t)3 * 256 * 128);
  cvt(mlp_w2, wmlp2h, (size_t)3 * 128 * 256);
  pack_convw_kernel<<<576, 256, 0, stream>>>(conv1_w, cw1h);
  pack_convw_kernel<<<576, 256, 0, stream>>>(conv2_w, cw2h);
  pack_convw_kernel<<<576, 256, 0, stream>>>(fus_w, fwh);

  const dim3 cgrid(2, 128, 12);

  // conv1 -> h1 (f16, relu), conv2 -> x_after (f32, relu)
  conv3x3_kernel<<<cgrid, 256, 0, stream>>>(xh, cw1h, conv1_b, nullptr, nullptr, nullptr, h1h, 1);
  conv3x3_kernel<<<cgrid, 256, 0, stream>>>(h1h, cw2h, conv2_b, nullptr, outx, nullptr, nullptr, 1);

  // rodin: means + fused 1x1 conv -> xr
  mean_axes_kernel<<<(6 * 128 * 128 + 255) / 256, 256, 0, stream>>>(outx, mean2b, mean3b);
  rodin_kernel<<<dim3(2, 128, 12), 256, 0, stream>>>(outx, mean2b, mean3b, rodin_w, rodin_b, xr, Bc);

  // tri-plane sampling -> c0 (f16)
  bilinear_kernel<<<Bc * Nc, 128, 0, stream>>>(xr, xyz, ybufh, Bc, Nc);

  // c = fc1_w2 @ relu(fc1_w1 @ c0) + fc2 @ c_res
  gemm_kernel<<<dim3(500, 4), 256, 0, stream>>>(ybufh, w11h, fc1_b1, nullptr, qkvb, qkvbh, 32000, 128, 256, 1);
  gemm_kernel<<<dim3(500, 2), 256, 0, stream>>>(cresh, w2h, fc2_b, nullptr, y2buf, nullptr, 32000, 128, 128, 0);
  gemm_kernel<<<dim3(500, 2), 256, 0, stream>>>(qkvbh, w12h, fc1_b2, y2buf, cbuf, nullptr, 32000, 256, 128, 0);

  // 3 windowed-attention blocks
  for (int i = 0; i < 3; ++i) {
    const int* idx  = indices     + (size_t)i * Bc * Nc;
    const int* idxi = indices_inv + (size_t)i * Bc * Nc;
    gather_kernel<<<Bc * Nc, 128, 0, stream>>>(cbuf, idx, cpbuf, Nc, 128);
    ln_kernel<<<4000, 256, 0, stream>>>(cpbuf, ln1_g + i * 128, ln1_b + i * 128, ybufh, 32000);
    gemm_kernel<<<dim3(500, 6), 256, 0, stream>>>(ybufh, wqkvh + (size_t)i * 384 * 128,
                                                  qkv_b + i * 384, nullptr, qkvb, nullptr, 32000, 128, 384, 0);
    attn_kernel<<<1024, 128, 0, stream>>>(qkvb, ybufh, Nc);
    gemm_kernel<<<dim3(500, 2), 256, 0, stream>>>(ybufh, wprojh + (size_t)i * 128 * 128,
                                                  proj_b + i * 128, cpbuf, y2buf, nullptr, 32000, 128, 128, 0);
    ln_kernel<<<4000, 256, 0, stream>>>(y2buf, ln2_g + i * 128, ln2_b + i * 128, ybufh, 32000);
    gemm_kernel<<<dim3(500, 4), 256, 0, stream>>>(ybufh, wmlp1h + (size_t)i * 256 * 128,
                                                  mlp_b1 + i * 256, nullptr, qkvb, qkvbh, 32000, 128, 256, 1);
    gemm_kernel<<<dim3(500, 2), 256, 0, stream>>>(qkvbh, wmlp2h + (size_t)i * 128 * 256,
                                                  mlp_b2 + i * 128, y2buf, cpbuf, nullptr, 32000, 256, 128, 0);
    float* dst = (i == 2) ? outc : cbuf;
    gather_kernel<<<Bc * Nc, 128, 0, stream>>>(cpbuf, idxi, dst, Nc, 128);
  }

  // scatter-mean over 3 plane projections -> hcat (f16)
  zero_kernel<<<(unsigned)((PLANE + 255) / 256), 256, 0, stream>>>(ssum, PLANE);
  zero_kernel<<<(6 * 16384 + 255) / 256, 256, 0, stream>>>(cnts, (size_t)6 * 16384);
  scatter_add_kernel<<<dim3(Bc * Nc, 3), 128, 0, stream>>>(outc, xyz, ssum, cnts, Bc, Nc);
  scatter_fin_kernel<<<(unsigned)((PLANE + 255) / 256), 256, 0, stream>>>(ssum, cnts, hcath);

  // xout = xr + fusion_conv(hcat); both output copies
  conv3x3_kernel<<<cgrid, 256, 0, stream>>>(hcath, fwh, fus_b, xr, out0, out1, nullptr, 0);
}
